// ImprovedReversibleQwen3CandidateAttention_1726576853572
// MI455X (gfx1250) — matrix operand order repaired and checked by reading, not compiled, not run on hardware
//
#include <hip/hip_runtime.h>
#include <hip/hip_bf16.h>

// ---------------------------------------------------------------------------
// Qwen3-style attention block for MI455X (gfx1250), wave32 + WMMA bf16.
// Pipeline: cast -> QKV WMMA GEMMs (async-LDS double-buffered, 128x256 tile,
// 64x64 per wave) -> RMSNorm+RoPE repack -> flash attention (async-LDS staged
// KV, WMMA QK^T/PV, uniform fast path off-diagonal) -> output WMMA GEMM.
// ---------------------------------------------------------------------------

typedef __attribute__((ext_vector_type(16))) __bf16 v16bf;
typedef __attribute__((ext_vector_type(8)))  __bf16 v8bf;
typedef __attribute__((ext_vector_type(8)))  float  v8f;
typedef unsigned short ushort_t;
typedef __attribute__((ext_vector_type(8))) unsigned short ushort8_t;

static constexpr int kS  = 2048;
static constexpr int kD  = 2048;
static constexpr int kH  = 16;
static constexpr int kKH = 8;
static constexpr int kHD = 128;
static constexpr int kNQ = kH * kHD;    // 2048
static constexpr int kNK = kKH * kHD;   // 1024
static constexpr float kEPS   = 1e-6f;
static constexpr float kScale = 0.08838834764831845f; // 1/sqrt(128)

// ---------------- helpers ----------------

__device__ __forceinline__ ushort_t f2bf(float f) {
  unsigned u = __builtin_bit_cast(unsigned, f);
  u += 0x7FFFu + ((u >> 16) & 1u);           // round-to-nearest-even
  return (ushort_t)(u >> 16);
}

__device__ __forceinline__ v8bf ld8bf(const ushort_t* p) {
  return __builtin_bit_cast(v8bf, *(const ushort8_t*)p);
}

__device__ __forceinline__ v16bf cat16(v8bf lo, v8bf hi) {
  v16bf r;
#pragma unroll
  for (int i = 0; i < 8; ++i) { r[i] = lo[i]; r[i + 8] = hi[i]; }
  return r;
}

__device__ __forceinline__ v8f wmma_bf16(v16bf a, v16bf b, v8f c) {
  return __builtin_amdgcn_wmma_f32_16x16x32_bf16(
      false, a, false, b, (short)0, c, false, false);
}

// Async global->LDS DMA (CDNA5): tracked with ASYNCcnt, lands directly in LDS.
__device__ __forceinline__ void async_ld_b128(const ushort_t* gaddr,
                                              const ushort_t* lds) {
  unsigned lds_off = (unsigned)(size_t)lds;    // flat LDS aperture: offset in [31:0]
  asm volatile("global_load_async_to_lds_b128 %0, %1, off"
               :: "v"(lds_off), "v"(gaddr) : "memory");
}
__device__ __forceinline__ void wait_async0() {
  asm volatile("s_wait_asynccnt 0" ::: "memory");
}

__device__ __forceinline__ float rmax16(float v) {
  v = fmaxf(v, __shfl_xor(v, 8, 32));
  v = fmaxf(v, __shfl_xor(v, 4, 32));
  v = fmaxf(v, __shfl_xor(v, 2, 32));
  v = fmaxf(v, __shfl_xor(v, 1, 32));
  return v;
}
__device__ __forceinline__ float rsum16(float v) {
  v += __shfl_xor(v, 8, 32);
  v += __shfl_xor(v, 4, 32);
  v += __shfl_xor(v, 2, 32);
  v += __shfl_xor(v, 1, 32);
  return v;
}
__device__ __forceinline__ float rsum32(float v) {
  v += __shfl_xor(v, 16, 32);
  return rsum16(v);
}

// ---------------- kernel 1: fp32 -> bf16 cast ----------------

__global__ void cast_f32_to_bf16(const float* __restrict__ in,
                                 ushort_t* __restrict__ out, int n4) {
  int i = blockIdx.x * blockDim.x + threadIdx.x;
  if (i < n4) {
    float4 v = ((const float4*)in)[i];
    ushort4 o;
    o.x = f2bf(v.x); o.y = f2bf(v.y); o.z = f2bf(v.z); o.w = f2bf(v.w);
    ((ushort4*)out)[i] = o;
  }
}

// ---------------- kernel 2: bf16 NT GEMM  C[M,N] = A[M,K] * B[N,K]^T ------
// 256 threads = 8 waves (2x4 wave grid); block tile 128x256, wave tile 64x64;
// BK=32; async-to-LDS double buffered (16 WMMA : 16 ds_load per k-step).

template <int N, int K>
__global__ __launch_bounds__(256)
void gemm_nt_bf16(const ushort_t* __restrict__ A,
                  const ushort_t* __restrict__ B,
                  float* __restrict__ C, int M) {
  constexpr int LDT = 40;                       // LDS row stride (ushorts)
  __shared__ ushort_t As[2][128 * LDT];         // 20 KB
  __shared__ ushort_t Bs[2][256 * LDT];         // 40 KB

  const int row0 = blockIdx.y * 128;
  const int col0 = blockIdx.x * 256;
  const int tid  = threadIdx.x;
  const int lane = tid & 31;
  const int w    = tid >> 5;
  const int wm   = w >> 2;                      // 0..1
  const int wn   = w & 3;                       // 0..3
  const int lr   = lane & 15;
  const int lh   = lane >> 4;
  const int ldr  = tid >> 2;                    // staging row 0..63 (+64..)
  const int seg  = tid & 3;                     // staging 16B segment

  auto stage = [&](int buf, int k0) {
#pragma unroll
    for (int it = 0; it < 2; ++it) {            // A: 128 rows x 4 segs
      int r = ldr + it * 64;
      async_ld_b128(&A[(size_t)(row0 + r) * K + k0 + seg * 8],
                    &As[buf][r * LDT + seg * 8]);
    }
#pragma unroll
    for (int it = 0; it < 4; ++it) {            // B: 256 rows x 4 segs
      int r = ldr + it * 64;
      async_ld_b128(&B[(size_t)(col0 + r) * K + k0 + seg * 8],
                    &Bs[buf][r * LDT + seg * 8]);
    }
  };

  v8f acc[4][4];
#pragma unroll
  for (int mi = 0; mi < 4; ++mi)
#pragma unroll
    for (int ni = 0; ni < 4; ++ni) acc[mi][ni] = (v8f)(0.0f);

  stage(0, 0);
  wait_async0();
  __syncthreads();

  int buf = 0;
#pragma unroll 1
  for (int k0 = 0; k0 < K; k0 += 32) {
    if (k0 + 32 < K) stage(buf ^ 1, k0 + 32);   // prefetch next tile

    v16bf af[4], bfv[4];
    const int kb = lh * 8;
#pragma unroll
    for (int mi = 0; mi < 4; ++mi) {
      int r = wm * 64 + mi * 16 + lr;
      af[mi] = cat16(ld8bf(&As[buf][r * LDT + kb]),
                     ld8bf(&As[buf][r * LDT + 16 + kb]));
    }
#pragma unroll
    for (int ni = 0; ni < 4; ++ni) {
      int r = wn * 64 + ni * 16 + lr;
      bfv[ni] = cat16(ld8bf(&Bs[buf][r * LDT + lh * 8]),
                      ld8bf(&Bs[buf][r * LDT + 16 + lh * 8]));
    }
#pragma unroll
    for (int mi = 0; mi < 4; ++mi)
#pragma unroll
      for (int ni = 0; ni < 4; ++ni)
        acc[mi][ni] = wmma_bf16(af[mi], bfv[ni], acc[mi][ni]);

    wait_async0();                              // next tile landed
    __syncthreads();                            // all waves done reading buf
    buf ^= 1;
  }

#pragma unroll
  for (int mi = 0; mi < 4; ++mi)
#pragma unroll
    for (int ni = 0; ni < 4; ++ni) {
      int mBase = row0 + wm * 64 + mi * 16 + 8 * lh;
      int n = col0 + wn * 64 + ni * 16 + lr;
      float* cp = &C[(size_t)mBase * N + n];
#pragma unroll
      for (int r = 0; r < 8; ++r) cp[r * N] = acc[mi][ni][r];
    }
}

// ---------------- kernel 3: RMSNorm + RoPE + head-major repack ------------
// One wave per (s, head) 128-vector. Q then K then V in vector index space.

__global__ __launch_bounds__(256)
void qk_norm_rope_v(const float* __restrict__ Qf, const float* __restrict__ Kf,
                    const float* __restrict__ Vf,
                    const float* __restrict__ cosr, const float* __restrict__ sinr,
                    const float* __restrict__ qw, const float* __restrict__ kw,
                    ushort_t* __restrict__ Qr, ushort_t* __restrict__ Kr,
                    ushort_t* __restrict__ Vt) {
  const int wid  = threadIdx.x >> 5;
  const int lane = threadIdx.x & 31;
  const int vec  = blockIdx.x * 8 + wid;
  const int d0   = lane * 4;
  const float sg = (lane < 16) ? -1.0f : 1.0f;   // rotate_half sign

  if (vec < kS * kH) {                            // ---- Q path
    int s = vec / kH, h = vec % kH;
    float4 t = *(const float4*)&Qf[(size_t)s * kNQ + h * kHD + d0];
    float ss = rsum32(t.x * t.x + t.y * t.y + t.z * t.z + t.w * t.w);
    float inv = rsqrtf(ss * (1.0f / kHD) + kEPS);
    float v[4] = {t.x, t.y, t.z, t.w};
    ushort4 o;
    ushort_t* op = (ushort_t*)&o;
    const float* cr = &cosr[(size_t)s * kHD + d0];
    const float* sr = &sinr[(size_t)s * kHD + d0];
#pragma unroll
    for (int j = 0; j < 4; ++j) {
      float tj  = v[j] * inv * qw[d0 + j];
      float oth = __shfl_xor(tj, 16, 32);        // value at d ^ 64
      op[j] = f2bf(tj * cr[j] + sg * oth * sr[j]);
    }
    *(ushort4*)&Qr[((size_t)h * kS + s) * kHD + d0] = o;
  } else if (vec < kS * (kH + kKH)) {             // ---- K path
    int i = vec - kS * kH;
    int s = i / kKH, h2 = i % kKH;
    float4 t = *(const float4*)&Kf[(size_t)s * kNK + h2 * kHD + d0];
    float ss = rsum32(t.x * t.x + t.y * t.y + t.z * t.z + t.w * t.w);
    float inv = rsqrtf(ss * (1.0f / kHD) + kEPS);
    float v[4] = {t.x, t.y, t.z, t.w};
    ushort4 o;
    ushort_t* op = (ushort_t*)&o;
    const float* cr = &cosr[(size_t)s * kHD + d0];
    const float* sr = &sinr[(size_t)s * kHD + d0];
#pragma unroll
    for (int j = 0; j < 4; ++j) {
      float tj  = v[j] * inv * kw[d0 + j];
      float oth = __shfl_xor(tj, 16, 32);
      op[j] = f2bf(tj * cr[j] + sg * oth * sr[j]);
    }
    *(ushort4*)&Kr[((size_t)h2 * kS + s) * kHD + d0] = o;
  } else {                                        // ---- V path (transpose)
    int i = vec - kS * (kH + kKH);
    int s = i / kKH, h2 = i % kKH;
    float4 t = *(const float4*)&Vf[(size_t)s * kNK + h2 * kHD + d0];
    float v[4] = {t.x, t.y, t.z, t.w};
#pragma unroll
    for (int j = 0; j < 4; ++j)
      Vt[((size_t)(h2 * kHD + d0 + j)) * kS + s] = f2bf(v[j]);
  }
}

// ---------------- kernel 4: causal flash attention (WMMA QK^T and PV) -----
// grid = H * S/64; 128 threads = 4 waves, each wave owns 16 q-rows.
// K/V chunks staged once per block via async-to-LDS (shared by all 4 waves).

__global__ __launch_bounds__(128)
void attention_causal(const ushort_t* __restrict__ Qr,
                      const ushort_t* __restrict__ Kr,
                      const ushort_t* __restrict__ Vt,
                      ushort_t* __restrict__ AO) {
  constexpr int KLD = 136;                       // K tile row stride (ushorts)
  constexpr int VLD = 40;                        // V tile row stride
  constexpr int PLD = 40;                        // per-wave P tile row stride
  __shared__ ushort_t Ks[32 * KLD];              // 32 keys x 128 d  (8704 B)
  __shared__ ushort_t Vs[128 * VLD];             // 128 d x 32 keys (10240 B)
  __shared__ ushort_t Pl[4 * 16 * PLD];          //                  (5120 B)

  const int tid  = threadIdx.x;
  const int lane = tid & 31;
  const int w    = tid >> 5;
  const int lr   = lane & 15;
  const int lh   = lane >> 4;
  const int h    = blockIdx.x >> 5;              // head
  const int qt   = blockIdx.x & 31;              // q tile
  const int q0   = qt * 64;
  const int kh2  = h >> 1;                       // GQA: g = H/KH = 2
  const int qrow0 = q0 + w * 16;

  const ushort_t* Kbase = Kr + (size_t)kh2 * kS * kHD;
  const ushort_t* Vbase = Vt + (size_t)kh2 * kHD * kS;

  // Q fragments (A layout): 4 frags cover d = 0..127.
  v16bf qf[4];
  {
    const ushort_t* qb = Qr + ((size_t)h * kS + qrow0 + lr) * kHD;
    const int kb = lh * 8;
#pragma unroll
    for (int di = 0; di < 4; ++di)
      qf[di] = cat16(ld8bf(&qb[di * 32 + kb]), ld8bf(&qb[di * 32 + 16 + kb]));
  }

  float mrow[8], lrow[8];
  v8f of[8];
#pragma unroll
  for (int r = 0; r < 8; ++r) { mrow[r] = -__builtin_inff(); lrow[r] = 0.0f; }
#pragma unroll
  for (int df = 0; df < 8; ++df) of[df] = (v8f)(0.0f);

  const int nk = (q0 + 64) >> 5;                 // 32-key chunks (causal bound)
#pragma unroll 1
  for (int c = 0; c < nk; ++c) {
    const int kb32 = c * 32;

    // --- cooperative async staging of K (32x64, d 0..63) and V (128x32) chunks
    __syncthreads();                             // prior reads of Ks/Vs done
#pragma unroll
    for (int it = 0; it < 2; ++it) {             // K: 256 x b128, 2/thread
      int idx = tid + it * 128;
      int row = idx >> 3, s8 = idx & 7;
      async_ld_b128(Kbase + (size_t)(kb32 + row) * kHD + s8 * 8,
                    &Ks[row * KLD + s8 * 8]);
    }
#pragma unroll
    for (int it = 0; it < 4; ++it) {             // V: 512 x b128, 4/thread
      int idx = tid + it * 128;
      int row = idx >> 2, s8 = idx & 3;
      async_ld_b128(Vbase + (size_t)row * kS + kb32 + s8 * 8,
                    &Vs[row * VLD + s8 * 8]);
    }
    wait_async0();
    __syncthreads();

    // --- scores: two 16-key fragments via WMMA over d (4 x K=32)
    v8f sfr[2];
#pragma unroll
    for (int hf = 0; hf < 2; ++hf) {
      const ushort_t* kp = &Ks[(hf * 16 + lr) * KLD + lh * 8];
      v8f sacc = (v8f)(0.0f);
#pragma unroll
      for (int di = 0; di < 4; ++di) {
        v16bf kf = cat16(ld8bf(&kp[di * 32]), ld8bf(&kp[di * 32 + 16]));
        sacc = wmma_bf16(qf[di], kf, sacc);
      }
      sfr[hf] = sacc;
    }

    if (kb32 + 31 <= qrow0) {
      // fast path: chunk strictly below the diagonal for every row of this
      // wave (wave-uniform, no barriers inside) -> no masking VALU.
#pragma unroll
      for (int hf = 0; hf < 2; ++hf)
#pragma unroll
        for (int r = 0; r < 8; ++r) sfr[hf][r] *= kScale;
    } else {
#pragma unroll
      for (int hf = 0; hf < 2; ++hf)
#pragma unroll
        for (int r = 0; r < 8; ++r) {
          int q = qrow0 + r + 8 * lh;
          float sv = sfr[hf][r] * kScale;
          sfr[hf][r] = (kb32 + hf * 16 + lr <= q) ? sv : -__builtin_inff();
        }
    }

    // --- online softmax row statistics
    float corr[8];
#pragma unroll
    for (int r = 0; r < 8; ++r) {
      float mx = rmax16(fmaxf(sfr[0][r], sfr[1][r]));
      float mnew = fmaxf(mrow[r], mx);
      corr[r] = __expf(mrow[r] - mnew);
      mrow[r] = mnew;
      float p0 = __expf(sfr[0][r] - mnew);
      float p1 = __expf(sfr[1][r] - mnew);
      sfr[0][r] = p0; sfr[1][r] = p1;
      lrow[r] = lrow[r] * corr[r] + rsum16(p0 + p1);
    }
#pragma unroll
    for (int df = 0; df < 8; ++df)
#pragma unroll
      for (int r = 0; r < 8; ++r) of[df][r] *= corr[r];

    // --- restage P (C/D layout) -> LDS -> A-fragment layout
#pragma unroll
    for (int hf = 0; hf < 2; ++hf)
#pragma unroll
      for (int r = 0; r < 8; ++r)
        Pl[w * 16 * PLD + (r + 8 * lh) * PLD + hf * 16 + lr] = f2bf(sfr[hf][r]);
    __syncthreads();

    v16bf pa;
    {
      const ushort_t* pb = &Pl[w * 16 * PLD + lr * PLD];
      const int kb = lh * 8;
      pa = cat16(ld8bf(&pb[kb]), ld8bf(&pb[16 + kb]));
    }

    // --- O += P @ V   (V transposed in LDS: contiguous in key dim)
#pragma unroll
    for (int df = 0; df < 8; ++df) {
      const ushort_t* vp = &Vs[(df * 16 + lr) * VLD + lh * 8];
      v16bf vf = cat16(ld8bf(vp), ld8bf(vp + 16));
      of[df] = wmma_bf16(pa, vf, of[df]);
    }
  }

  // --- normalize and write out (row-major [s, h*128] bf16 for out-proj)
  float linv[8];
#pragma unroll
  for (int r = 0; r < 8; ++r) linv[r] = 1.0f / lrow[r];
#pragma unroll
  for (int df = 0; df < 8; ++df) {
    int col = h * kHD + df * 16 + lr;
#pragma unroll
    for (int r = 0; r < 8; ++r) {
      int row = qrow0 + r + 8 * lh;
      AO[(size_t)row * kNQ + col] = f2bf(of[df][r] * linv[r]);
    }
  }
}

// ---------------- host side ----------------

extern "C" void kernel_launch(void* const* d_in, const int* in_sizes, int n_in,
                              void* d_out, int out_size, void* d_ws, size_t ws_size,
                              hipStream_t stream) {
  (void)in_sizes; (void)n_in; (void)out_size; (void)ws_size;
  const float* x    = (const float*)d_in[0];
  const float* cosr = (const float*)d_in[1];
  const float* sinr = (const float*)d_in[2];
  const float* wq   = (const float*)d_in[3];
  const float* wk   = (const float*)d_in[4];
  const float* wv   = (const float*)d_in[5];
  const float* wo   = (const float*)d_in[6];
  const float* qnw  = (const float*)d_in[7];
  const float* knw  = (const float*)d_in[8];
  float* out = (float*)d_out;

  char* ws = (char*)d_ws;
  size_t off = 0;
  auto alloc = [&](size_t bytes) -> void* {
    void* p = ws + off;
    off += (bytes + 255) & ~(size_t)255;
    return p;
  };
  ushort_t* xb  = (ushort_t*)alloc((size_t)kS * kD * 2);
  ushort_t* wqb = (ushort_t*)alloc((size_t)kNQ * kD * 2);
  ushort_t* wkb = (ushort_t*)alloc((size_t)kNK * kD * 2);
  ushort_t* wvb = (ushort_t*)alloc((size_t)kNK * kD * 2);
  ushort_t* wob = (ushort_t*)alloc((size_t)kD * kNQ * 2);
  float*    Qf  = (float*)alloc((size_t)kS * kNQ * 4);
  float*    Kf  = (float*)alloc((size_t)kS * kNK * 4);
  float*    Vf  = (float*)alloc((size_t)kS * kNK * 4);
  ushort_t* Qr  = (ushort_t*)alloc((size_t)kH * kS * kHD * 2);
  ushort_t* Kr  = (ushort_t*)alloc((size_t)kKH * kS * kHD * 2);
  ushort_t* Vt  = (ushort_t*)alloc((size_t)kKH * kHD * kS * 2);
  ushort_t* AO  = (ushort_t*)alloc((size_t)kS * kNQ * 2);

  auto cast = [&](const float* src, ushort_t* dst, int n) {
    int n4 = n / 4;
    cast_f32_to_bf16<<<(n4 + 255) / 256, 256, 0, stream>>>(src, dst, n4);
  };
  cast(x,  xb,  kS * kD);
  cast(wq, wqb, kNQ * kD);
  cast(wk, wkb, kNK * kD);
  cast(wv, wvb, kNK * kD);
  cast(wo, wob, kD * kNQ);

  gemm_nt_bf16<kNQ, kD><<<dim3(kNQ / 256, kS / 128), 256, 0, stream>>>(xb, wqb, Qf, kS);
  gemm_nt_bf16<kNK, kD><<<dim3(kNK / 256, kS / 128), 256, 0, stream>>>(xb, wkb, Kf, kS);
  gemm_nt_bf16<kNK, kD><<<dim3(kNK / 256, kS / 128), 256, 0, stream>>>(xb, wvb, Vf, kS);

  qk_norm_rope_v<<<(kS * (kH + 2 * kKH)) / 8, 256, 0, stream>>>(
      Qf, Kf, Vf, cosr, sinr, qnw, knw, Qr, Kr, Vt);

  attention_causal<<<kH * (kS / 64), 128, 0, stream>>>(Qr, Kr, Vt, AO);

  gemm_nt_bf16<kD, kNQ><<<dim3(kD / 256, kS / 128), 256, 0, stream>>>(AO, wob, out, kS);
}
